// SimpleAdapterModel_40501541601446
// MI455X (gfx1250) — matrix operand order repaired and checked by reading, not compiled, not run on hardware
//
#include <hip/hip_runtime.h>

typedef __attribute__((ext_vector_type(16))) __bf16 v16bf;
typedef __attribute__((ext_vector_type(8)))  __bf16 v8bf;
typedef __attribute__((ext_vector_type(8)))  float  v8f;

#define CAT16(lo, hi) __builtin_shufflevector((lo), (hi), 0,1,2,3,4,5,6,7,8,9,10,11,12,13,14,15)

// Problem constants (match reference)
constexpr int NB   = 256;
constexpr int NP   = 128;
constexpr int NTOK = NB * NP;    // 32768
constexpr int D    = 768;
constexpr int E    = 64;
constexpr int H    = 256;
constexpr int H2   = 128;
constexpr int O    = 2;

constexpr int MT       = 32;                  // rows (tokens) per block tile
constexpr int MAXTILES = NTOK / MT + E;       // 1088

// LDS strides (bf16 elements), padded to spread banks
constexpr int ASTRIDE  = D  + 8;              // 776 -> 1552 B (== 4 banks mod 64)
constexpr int H1STRIDE = H  + 8;              // 264
constexpr int H2STRIDE = H2 + 8;              // 136

// Header layout (ints) inside workspace
constexpr int HDR_COUNTS  = 0;      // [64]
constexpr int HDR_OFFSETS = 64;     // [64]
constexpr int HDR_CURSORS = 128;    // [64]
constexpr int HDR_NTILES  = 192;    // [1]
constexpr int HDR_TILE_E  = 256;    // [4096]
constexpr int HDR_TILE_R0 = 4352;   // [4096]
constexpr int HDR_INTS    = 16384;  // 64 KB header

// ---------------------------------------------------------------- init ----
__global__ void k_init(float* __restrict__ out, int n_out, int* __restrict__ hdr) {
    int i = blockIdx.x * blockDim.x + threadIdx.x;
    if (i < n_out)    out[i] = 0.0f;
    if (i < HDR_INTS) hdr[i] = 0;
}

// --------------------------------------------------------------- count ----
__global__ void k_count(const int* __restrict__ prop, const float* __restrict__ mask,
                        int* __restrict__ hdr) {
    int i = blockIdx.x * blockDim.x + threadIdx.x;
    if (i < NTOK && mask[i] > 0.0f) atomicAdd(&hdr[HDR_COUNTS + prop[i]], 1);
}

// ---------------------------------------------------------------- scan ----
__global__ void k_scan(int* __restrict__ hdr) {
    if (threadIdx.x != 0) return;
    int off = 0, nt = 0;
    for (int e = 0; e < E; ++e) {
        int c = hdr[HDR_COUNTS + e];
        hdr[HDR_OFFSETS + e] = off;
        int tiles = (c + MT - 1) / MT;
        for (int i = 0; i < tiles; ++i) {
            hdr[HDR_TILE_E  + nt] = e;
            hdr[HDR_TILE_R0 + nt] = off + i * MT;
            ++nt;
        }
        off += c;
    }
    hdr[HDR_NTILES] = nt;
}

// -------------------------------------------- LayerNorm + bf16 scatter ----
__global__ __launch_bounds__(256)
void k_ln_scatter(const float* __restrict__ x, const int* __restrict__ prop,
                  const float* __restrict__ mask, const float* __restrict__ g,
                  const float* __restrict__ bt, int* __restrict__ hdr,
                  int* __restrict__ token_id, __bf16* __restrict__ Y) {
    const int tok = blockIdx.x;
    const int tid = threadIdx.x;
    const float* xr = x + (size_t)tok * D;
    float v0 = xr[tid], v1 = xr[tid + 256], v2 = xr[tid + 512];

    __shared__ float rs_[256], rq_[256];
    rs_[tid] = v0 + v1 + v2;
    rq_[tid] = v0 * v0 + v1 * v1 + v2 * v2;
    __syncthreads();
    for (int w = 128; w > 0; w >>= 1) {
        if (tid < w) { rs_[tid] += rs_[tid + w]; rq_[tid] += rq_[tid + w]; }
        __syncthreads();
    }
    const float mu  = rs_[0] * (1.0f / 768.0f);
    const float var = rq_[0] * (1.0f / 768.0f) - mu * mu;
    const float ri  = rsqrtf(var + 1e-5f);

    if (mask[tok] <= 0.0f) return;          // uniform per block
    const int e = prop[tok];

    __shared__ int slot_s;
    if (tid == 0) {
        int slot = hdr[HDR_OFFSETS + e] + atomicAdd(&hdr[HDR_CURSORS + e], 1);
        slot_s = slot;
        token_id[slot] = tok;
    }
    __syncthreads();
    const int slot = slot_s;

    __bf16* yr = Y + (size_t)slot * D;
    const float* ge = g  + (size_t)e * D;
    const float* be = bt + (size_t)e * D;
    yr[tid      ] = (__bf16)((v0 - mu) * ri * ge[tid      ] + be[tid      ]);
    yr[tid + 256] = (__bf16)((v1 - mu) * ri * ge[tid + 256] + be[tid + 256]);
    yr[tid + 512] = (__bf16)((v2 - mu) * ri * ge[tid + 512] + be[tid + 512]);
}

// --------------------------- transpose fp32 [E][K][N] -> bf16 [E][N][K] ----
__global__ __launch_bounds__(256)
void k_transpose_cvt(const float* __restrict__ src, __bf16* __restrict__ dst,
                     int K, int N) {
    const int tilesN = N >> 5, tilesK = K >> 5;
    const int b  = blockIdx.x;
    const int e  = b / (tilesK * tilesN);
    const int rm = b % (tilesK * tilesN);
    const int kt = rm / tilesN, nt = rm % tilesN;
    const float* s = src + (size_t)e * K * N;
    __bf16*      d = dst + (size_t)e * K * N;

    __shared__ float tile[32][33];
    const int tid = threadIdx.x;
    const int c = tid & 31, r = tid >> 5;   // r in 0..7
    for (int i = 0; i < 4; ++i)
        tile[r + i * 8][c] = s[(size_t)(kt * 32 + r + i * 8) * N + nt * 32 + c];
    __syncthreads();
    for (int i = 0; i < 4; ++i)
        d[(size_t)(nt * 32 + r + i * 8) * K + kt * 32 + c] = (__bf16)tile[c][r + i * 8];
}

// ------------------------------------------------- grouped-GEMM MLP -------
__global__ __launch_bounds__(256)
void k_mlp(const __bf16* __restrict__ Y, const __bf16* __restrict__ W1t,
           const __bf16* __restrict__ W2t, const float* __restrict__ b1,
           const float* __restrict__ b2, const float* __restrict__ W3,
           const float* __restrict__ b3, const float* __restrict__ base,
           const float* __restrict__ mask, const int* __restrict__ hdr,
           const int* __restrict__ token_id, float* __restrict__ out) {
    const int t = blockIdx.x;
    if (t >= hdr[HDR_NTILES]) return;
    const int e  = hdr[HDR_TILE_E + t];
    const int r0 = hdr[HDR_TILE_R0 + t];
    const int rows_valid = min(MT, hdr[HDR_OFFSETS + e] + hdr[HDR_COUNTS + e] - r0);

    __shared__ __align__(16) __bf16 Alds[MT][ASTRIDE];      // 48.5 KB activation tile
    __shared__ __align__(16) __bf16 h1s[MT][H1STRIDE];
    __shared__ __align__(16) __bf16 h2s[MT][H2STRIDE];

    const int tid  = threadIdx.x;
    const int wave = tid >> 5;
    const int lane = tid & 31;
    const int lrow = lane & 15;
    const int lhi  = lane >> 4;

    // -------- async-stage A tile (MT x D bf16) into LDS ------------------
    // Each thread copies 12 x 16B chunks: row = tid>>3, cols (tid&7)+8j.
    {
        const int row = tid >> 3;
        const char* grow = (const char*)(Y + (size_t)(r0 + row) * D);
        unsigned int lrowoff =
            (unsigned int)(size_t)&Alds[row][0];           // low 32 bits = LDS offset
        #pragma unroll
        for (int j = 0; j < 12; ++j) {
            const int cb = ((tid & 7) + j * 8) * 16;       // byte offset in row
            unsigned long long ga = (unsigned long long)(grow + cb);
            unsigned int la = lrowoff + (unsigned int)cb;
            asm volatile("global_load_async_to_lds_b128 %0, %1, off"
                         :: "v"(la), "v"(ga) : "memory");
        }
        asm volatile("s_wait_asynccnt 0x0" ::: "memory");
    }
    __syncthreads();

    // ---------------- layer 1: [MTxD] @ [DxH], bf16 WMMA, f32 acc --------
    const __bf16* W1e  = W1t + (size_t)e * H * D;
    const int n0 = wave * 2, n1 = n0 + 1;
    const __bf16* B0row = W1e + (size_t)(n0 * 16 + lrow) * D + lhi * 8;
    const __bf16* B1row = W1e + (size_t)(n1 * 16 + lrow) * D + lhi * 8;

    v8f c00 = {}; v8f c01 = {}; v8f c10 = {}; v8f c11 = {};
    for (int k0 = 0; k0 < D; k0 += 32) {
        const int ab = k0 + lhi * 8;
        v8bf a0lo = *(const v8bf*)&Alds[lrow     ][ab];
        v8bf a0hi = *(const v8bf*)&Alds[lrow     ][ab + 16];
        v8bf a1lo = *(const v8bf*)&Alds[lrow + 16][ab];
        v8bf a1hi = *(const v8bf*)&Alds[lrow + 16][ab + 16];
        v16bf a0 = CAT16(a0lo, a0hi);
        v16bf a1 = CAT16(a1lo, a1hi);
        v8bf b0lo = *(const v8bf*)(B0row + k0);
        v8bf b0hi = *(const v8bf*)(B0row + k0 + 16);
        v16bf bb0 = CAT16(b0lo, b0hi);
        v8bf b1lo = *(const v8bf*)(B1row + k0);
        v8bf b1hi = *(const v8bf*)(B1row + k0 + 16);
        v16bf bb1 = CAT16(b1lo, b1hi);
        c00 = __builtin_amdgcn_wmma_f32_16x16x32_bf16(false, a0, false, bb0, (short)0, c00, false, false);
        c01 = __builtin_amdgcn_wmma_f32_16x16x32_bf16(false, a0, false, bb1, (short)0, c01, false, false);
        c10 = __builtin_amdgcn_wmma_f32_16x16x32_bf16(false, a1, false, bb0, (short)0, c10, false, false);
        c11 = __builtin_amdgcn_wmma_f32_16x16x32_bf16(false, a1, false, bb1, (short)0, c11, false, false);
    }
    {
        const float bias0 = b1[e * H + n0 * 16 + lrow];
        const float bias1 = b1[e * H + n1 * 16 + lrow];
        #pragma unroll
        for (int v = 0; v < 8; ++v) {
            const int row = v + lhi * 8;
            float x;
            x = c00[v] + bias0; x = x > 0.0f ? x : 0.0f; h1s[row     ][n0 * 16 + lrow] = (__bf16)x;
            x = c01[v] + bias1; x = x > 0.0f ? x : 0.0f; h1s[row     ][n1 * 16 + lrow] = (__bf16)x;
            x = c10[v] + bias0; x = x > 0.0f ? x : 0.0f; h1s[row + 16][n0 * 16 + lrow] = (__bf16)x;
            x = c11[v] + bias1; x = x > 0.0f ? x : 0.0f; h1s[row + 16][n1 * 16 + lrow] = (__bf16)x;
        }
    }
    __syncthreads();

    // ---------------- layer 2: [MTxH] @ [HxH2] ---------------------------
    const __bf16* W2e  = W2t + (size_t)e * H2 * H;
    const int nn = wave;
    const __bf16* Brow = W2e + (size_t)(nn * 16 + lrow) * H + lhi * 8;
    v8f d0 = {}; v8f d1 = {};
    for (int k0 = 0; k0 < H; k0 += 32) {
        const int ab = k0 + lhi * 8;
        v8bf a0lo = *(const v8bf*)&h1s[lrow     ][ab];
        v8bf a0hi = *(const v8bf*)&h1s[lrow     ][ab + 16];
        v8bf a1lo = *(const v8bf*)&h1s[lrow + 16][ab];
        v8bf a1hi = *(const v8bf*)&h1s[lrow + 16][ab + 16];
        v16bf a0 = CAT16(a0lo, a0hi);
        v16bf a1 = CAT16(a1lo, a1hi);
        v8bf blo = *(const v8bf*)(Brow + k0);
        v8bf bhi = *(const v8bf*)(Brow + k0 + 16);
        v16bf bb = CAT16(blo, bhi);
        d0 = __builtin_amdgcn_wmma_f32_16x16x32_bf16(false, a0, false, bb, (short)0, d0, false, false);
        d1 = __builtin_amdgcn_wmma_f32_16x16x32_bf16(false, a1, false, bb, (short)0, d1, false, false);
    }
    {
        const float bias = b2[e * H2 + nn * 16 + lrow];
        #pragma unroll
        for (int v = 0; v < 8; ++v) {
            const int row = v + lhi * 8;
            float x;
            x = d0[v] + bias; x = x > 0.0f ? x : 0.0f; h2s[row     ][nn * 16 + lrow] = (__bf16)x;
            x = d1[v] + bias; x = x > 0.0f ? x : 0.0f; h2s[row + 16][nn * 16 + lrow] = (__bf16)x;
        }
    }
    __syncthreads();

    // ---------------- layer 3 (H2 -> O=2) + residual + scatter -----------
    if (tid < 64) {
        const int row = tid & 31;
        const int o   = tid >> 5;
        const float* w3 = W3 + (size_t)e * H2 * O + o;
        float s = 0.0f;
        for (int k = 0; k < H2; ++k) s += (float)h2s[row][k] * w3[k * O];
        s += b3[e * O + o];
        if (row < rows_valid) {
            const int tok = token_id[r0 + row];
            const float m = mask[tok];
            out[(size_t)tok * O + o] =
                (0.7f * s + 0.3f * base[(size_t)tok * O + o]) * m;
        }
    }
}

// ------------------------------------------------------------ launcher ----
extern "C" void kernel_launch(void* const* d_in, const int* in_sizes, int n_in,
                              void* d_out, int out_size, void* d_ws, size_t ws_size,
                              hipStream_t stream) {
    (void)in_sizes; (void)n_in; (void)ws_size;
    const float* hidden = (const float*)d_in[0];
    const float* basep  = (const float*)d_in[1];
    const int*   prop   = (const int*)  d_in[2];
    const float* maskp  = (const float*)d_in[3];
    const float* ln_g   = (const float*)d_in[4];
    const float* ln_b   = (const float*)d_in[5];
    const float* W1     = (const float*)d_in[6];
    const float* b1     = (const float*)d_in[7];
    const float* W2     = (const float*)d_in[8];
    const float* b2     = (const float*)d_in[9];
    const float* W3     = (const float*)d_in[10];
    const float* b3     = (const float*)d_in[11];
    float* out = (float*)d_out;

    // Workspace layout (256B aligned regions)
    char* ws = (char*)d_ws;
    int*    hdr      = (int*)ws;                                    // 64 KB
    int*    token_id = (int*)(ws + 65536);                          // 128 KB
    __bf16* Yb       = (__bf16*)(ws + 65536 + 131072);              // (N+MT)*D*2
    size_t y_bytes   = (size_t)(NTOK + MT) * D * sizeof(__bf16);
    __bf16* W1t      = (__bf16*)(ws + 65536 + 131072 + y_bytes);
    size_t w1_bytes  = (size_t)E * H * D * sizeof(__bf16);
    __bf16* W2t      = (__bf16*)((char*)W1t + w1_bytes);

    k_init<<<(HDR_INTS + 255) / 256, 256, 0, stream>>>(out, out_size, hdr);
    k_count<<<NTOK / 256, 256, 0, stream>>>(prop, maskp, hdr);
    k_scan<<<1, 64, 0, stream>>>(hdr);
    k_transpose_cvt<<<E * (D / 32) * (H / 32), 256, 0, stream>>>(W1, W1t, D, H);
    k_transpose_cvt<<<E * (H / 32) * (H2 / 32), 256, 0, stream>>>(W2, W2t, H, H2);
    k_ln_scatter<<<NTOK, 256, 0, stream>>>(hidden, prop, maskp, ln_g, ln_b,
                                           hdr, token_id, Yb);
    k_mlp<<<MAXTILES, 256, 0, stream>>>(Yb, W1t, W2t, b1, b2, W3, b3,
                                        basep, maskp, hdr, token_id, out);
}
